// MSDeformAttn_24859270709854
// MI455X (gfx1250) — matrix operand order repaired and checked by reading, not compiled, not run on hardware
//
#include <hip/hip_runtime.h>
#include <hip/hip_bf16.h>

#define NH_ 8
#define NL_ 3
#define NP_ 4
#define HD_ 32
#define B_  2
#define LV_ 11109
#define LQ_ 11109
#define MROWS (B_ * LQ_)   /* 22218 */

typedef __attribute__((ext_vector_type(16))) __bf16        v16bf;
typedef __attribute__((ext_vector_type(8)))  float         v8f;
typedef __attribute__((ext_vector_type(8)))  unsigned int  v8u;
typedef __attribute__((ext_vector_type(4)))  unsigned int  u32x4;
typedef __attribute__((ext_vector_type(8)))  unsigned int  u32x8;

__device__ __forceinline__ unsigned short f2bf(float f) {
    unsigned int u = __float_as_uint(f);
    unsigned int r = u + 0x7FFFu + ((u >> 16) & 1u);   // round-to-nearest-even
    return (unsigned short)(r >> 16);
}

// ---------------------------------------------------------------------------
// C[M,N] = A[M,256] @ W[256,N] + bias[N]   (fp32 in/out, bf16 WMMA internal)
//
// Block: 256 threads = 8 waves. Workgroup tile 128(M) x 64(N).
// Wave (wm = w&3, wn = w>>2) owns a 32x32 slab -> 2 A-frags x 2 B-frags =
// 4 x v_wmma_f32_16x16x32_bf16 per 32-wide K step, 8 K steps.
//
// A tile (128x32) is staged by all threads (f32 -> bf16 in LDS).
// B tile (32x64 f32) is staged by the Tensor Data Mover: wave 0 issues
// tensor_load_to_lds with a 2-group D# descriptor, waits TENSORcnt, barrier.
// ---------------------------------------------------------------------------
__global__ __launch_bounds__(256)
void gemm_bias_wmma(const float* __restrict__ A, const float* __restrict__ Wt,
                    const float* __restrict__ bias, float* __restrict__ C,
                    int M, int N) {
    __shared__ __align__(16) unsigned short As[128][34];  // [m][k] bf16, padded
    __shared__ __align__(16) float          BsF[32][64];  // [k][n] f32 (TDM image)

    const int t     = threadIdx.x;
    const int wave  = t >> 5;
    const int lane  = t & 31;
    const int wm    = wave & 3;           // 0..3  -> rows wm*32
    const int wn    = wave >> 2;          // 0..1  -> cols wn*32
    const int col16 = lane & 15;
    const int half  = lane >> 4;
    const int blockM = blockIdx.y * 128;
    const int blockN = blockIdx.x * 64;

    // TDM D# pieces that do not change across the K loop (wave-uniform)
    const unsigned lds_b   = (unsigned)(size_t)&BsF[0][0];       // LDS byte offset
    const unsigned remN    = (unsigned)(N - blockN);             // OOB width from tile start
    const unsigned tdim1   = 256u;                               // rows remaining (always >=32)

    v8f acc00 = {}, acc01 = {}, acc10 = {}, acc11 = {};

    for (int kb = 0; kb < 256; kb += 32) {
        // ---- stage A tile (128x32 f32 -> bf16), coalesced float4 loads ----
        #pragma unroll
        for (int i = 0; i < 4; ++i) {
            int idx  = t + i * 256;            // 0..1023
            int row  = idx >> 3;               // 0..127
            int col4 = (idx & 7) << 2;         // 0..28
            int gr   = blockM + row;
            float4 v = make_float4(0.f, 0.f, 0.f, 0.f);
            if (gr < M) v = *(const float4*)&A[(size_t)gr * 256 + kb + col4];
            As[row][col4 + 0] = f2bf(v.x);
            As[row][col4 + 1] = f2bf(v.y);
            As[row][col4 + 2] = f2bf(v.z);
            As[row][col4 + 3] = f2bf(v.w);
        }

        // ---- stage W tile (32 x 64 f32) via Tensor Data Mover (wave 0) ----
        if (wave == 0) {
            unsigned long long gaddr =
                (unsigned long long)(const void*)(Wt + (size_t)kb * N + blockN);
            u32x4 g0;
            g0[0] = 1u;                                       // count=1, user mode
            g0[1] = lds_b;                                    // lds_addr (bytes)
            g0[2] = (unsigned)(gaddr & 0xffffffffu);          // global_addr[31:0]
            g0[3] = (unsigned)((gaddr >> 32) & 0x1ffffffu)    // global_addr[56:32]
                  | (2u << 30);                               // type = 2 (image)
            u32x8 g1;
            g1[0] = (2u << 16);                               // wg_mask=0, data_size=4B
            g1[1] = (remN & 0xffffu) << 16;                   // tensor_dim0[15:0]
            g1[2] = (remN >> 16) | ((tdim1 & 0xffffu) << 16); // dim0 hi | tensor_dim1 lo
            g1[3] = (tdim1 >> 16) | (64u << 16);              // dim1 hi | tile_dim0 = 64
            g1[4] = 32u;                                      // tile_dim1 = 32, tile_dim2 = 0
            g1[5] = (unsigned)N;                              // tensor_dim0_stride[31:0]
            g1[6] = 0u;                                       // stride hi | dim1_stride lo
            g1[7] = 0u;
            asm volatile("tensor_load_to_lds %0, %1"
                         :: "s"(g0), "s"(g1) : "memory");
            __builtin_amdgcn_s_wait_tensorcnt(0);
        }
        __syncthreads();

        // ---- build fragments per ISA 16-bit layouts ----
        v8u au0, au1, bu0, bu1;
        const int arow0 = wm * 32 + col16;
        const int arow1 = arow0 + 16;
        const int bcol0 = wn * 32 + col16;
        const int bcol1 = bcol0 + 16;
        #pragma unroll
        for (int v = 0; v < 8; ++v) {
            int ka = ((v < 4) ? 0 : 16) + half * 8 + (v & 3) * 2; // A: lanes16-31 = K+8
            au0[v] = *(const unsigned int*)&As[arow0][ka];
            au1[v] = *(const unsigned int*)&As[arow1][ka];
            int kv = ((v < 4) ? 0 : 16) + half * 8 + (v & 3) * 2; // B: lanes16-31 = K+8
            float f0 = BsF[kv][bcol0], f1 = BsF[kv + 1][bcol0];
            bu0[v] = (unsigned)f2bf(f0) | ((unsigned)f2bf(f1) << 16);
            float g0f = BsF[kv][bcol1], g1f = BsF[kv + 1][bcol1];
            bu1[v] = (unsigned)f2bf(g0f) | ((unsigned)f2bf(g1f) << 16);
        }
        v16bf a0 = __builtin_bit_cast(v16bf, au0);
        v16bf a1 = __builtin_bit_cast(v16bf, au1);
        v16bf b0 = __builtin_bit_cast(v16bf, bu0);
        v16bf b1 = __builtin_bit_cast(v16bf, bu1);
        acc00 = __builtin_amdgcn_wmma_f32_16x16x32_bf16(false, a0, false, b0,
                                                        (short)0, acc00, false, false);
        acc01 = __builtin_amdgcn_wmma_f32_16x16x32_bf16(false, a0, false, b1,
                                                        (short)0, acc01, false, false);
        acc10 = __builtin_amdgcn_wmma_f32_16x16x32_bf16(false, a1, false, b0,
                                                        (short)0, acc10, false, false);
        acc11 = __builtin_amdgcn_wmma_f32_16x16x32_bf16(false, a1, false, b1,
                                                        (short)0, acc11, false, false);
        __syncthreads();
    }

    // ---- store: C layout VGPR r -> row r (lanes 0-15) / r+8 (lanes 16-31) ----
    const int gc0 = blockN + wn * 32 + col16;
    const int gc1 = gc0 + 16;
    #pragma unroll
    for (int r = 0; r < 8; ++r) {
        int row0 = blockM + wm * 32 + r + half * 8;
        int row1 = row0 + 16;
        if (row0 < M) {
            if (gc0 < N) C[(size_t)row0 * N + gc0] = acc00[r] + bias[gc0];
            if (gc1 < N) C[(size_t)row0 * N + gc1] = acc01[r] + bias[gc1];
        }
        if (row1 < M) {
            if (gc0 < N) C[(size_t)row1 * N + gc0] = acc10[r] + bias[gc0];
            if (gc1 < N) C[(size_t)row1 * N + gc1] = acc11[r] + bias[gc1];
        }
    }
}

// ---------------------------------------------------------------------------
// Deformable sampling: softmax(12) + 12 bilinear gathers per (b,q,head).
// Block (32 channels, 8 heads); grid = B*LQ.
// ---------------------------------------------------------------------------
__global__ __launch_bounds__(256)
void msda_sample(const float* __restrict__ value,   // [B*LV, 256]
                 const float* __restrict__ off,     // [B*LQ, 192] (h,l,p,(dy,dx))
                 const float* __restrict__ logits,  // [B*LQ, 96]  (h,l,p)
                 const float* __restrict__ refp,    // [B*LQ, 3, 2] (y,x)
                 float* __restrict__ mid)           // [B*LQ, 256]
{
    constexpr int LVL_H[3] = {92, 46, 23};
    constexpr int LVL_W[3] = {92, 46, 23};
    constexpr int LVL_S[3] = {0, 8464, 10580};

    const int bq = blockIdx.x;          // 0..MROWS-1
    const int h  = threadIdx.y;         // 0..7
    const int c  = threadIdx.x;         // 0..31
    const int b  = bq / LQ_;

    // per-(b,q,h) softmax over 12 logits (uniform across the 32 lanes)
    const float* lg = logits + (size_t)bq * 96 + h * 12;
    float w[12];
    float mx = -1e30f;
    #pragma unroll
    for (int j = 0; j < 12; ++j) { w[j] = lg[j]; mx = fmaxf(mx, w[j]); }
    float s = 0.f;
    #pragma unroll
    for (int j = 0; j < 12; ++j) { w[j] = __expf(w[j] - mx); s += w[j]; }
    const float inv = 1.f / s;

    const float* op = off  + (size_t)bq * 192 + h * 24;
    const float* rp = refp + (size_t)bq * 6;

    float acc = 0.f;
    #pragma unroll
    for (int l = 0; l < NL_; ++l) {
        const int   H  = LVL_H[l], W = LVL_W[l];
        const float ry = rp[l * 2 + 0];
        const float rx = rp[l * 2 + 1];
        const float* vbase = value + ((size_t)(b * LV_ + LVL_S[l])) * 256 + h * 32 + c;
        #pragma unroll
        for (int p = 0; p < NP_; ++p) {
            float dy = op[(l * 4 + p) * 2 + 0];
            float dx = op[(l * 4 + p) * 2 + 1];
            // grid_sample(align_corners=False): pix = loc*dim - 0.5
            float y = (ry + dy / (float)H) * (float)H - 0.5f;
            float x = (rx + dx / (float)W) * (float)W - 0.5f;
            float y0f = floorf(y), x0f = floorf(x);
            float ly = y - y0f, lx = x - x0f;
            int y0 = (int)y0f, x0 = (int)x0f;
            bool xi0 = (x0 >= 0)     && (x0 < W);
            bool xi1 = (x0 + 1 >= 0) && (x0 + 1 < W);
            bool yi0 = (y0 >= 0)     && (y0 < H);
            bool yi1 = (y0 + 1 >= 0) && (y0 + 1 < H);
            float v00 = 0.f, v01 = 0.f, v10 = 0.f, v11 = 0.f;
            if (yi0 && xi0) v00 = vbase[(size_t)(y0 * W + x0)       * 256];
            if (yi0 && xi1) v01 = vbase[(size_t)(y0 * W + x0 + 1)   * 256];
            if (yi1 && xi0) v10 = vbase[(size_t)((y0 + 1) * W + x0) * 256];
            if (yi1 && xi1) v11 = vbase[(size_t)((y0 + 1) * W + x0 + 1) * 256];
            float bil = v00 * (1.f - ly) * (1.f - lx) + v01 * (1.f - ly) * lx
                      + v10 * ly * (1.f - lx)         + v11 * ly * lx;
            acc += w[l * 4 + p] * inv * bil;
        }
    }
    mid[(size_t)bq * 256 + h * 32 + c] = acc;
}

// ---------------------------------------------------------------------------
extern "C" void kernel_launch(void* const* d_in, const int* in_sizes, int n_in,
                              void* d_out, int out_size, void* d_ws, size_t ws_size,
                              hipStream_t stream) {
    const float* query  = (const float*)d_in[0];
    const float* refp   = (const float*)d_in[1];
    const float* vflat  = (const float*)d_in[2];
    const float* W_val  = (const float*)d_in[3];
    const float* b_val  = (const float*)d_in[4];
    const float* W_off  = (const float*)d_in[5];
    const float* b_off  = (const float*)d_in[6];
    const float* W_attn = (const float*)d_in[7];
    const float* b_attn = (const float*)d_in[8];
    const float* W_out  = (const float*)d_in[9];
    const float* b_out  = (const float*)d_in[10];
    float* out = (float*)d_out;
    float* ws  = (float*)d_ws;

    const int M  = MROWS;                        // 22218
    const int mt = (M + 127) / 128;              // 174 M-tiles

    float* ws_val = ws;                          // [M,256]
    float* ws_off = ws_val + (size_t)M * 256;    // [M,192]
    float* ws_log = ws_off + (size_t)M * 192;    // [M, 96]
    float* ws_mid = ws_log + (size_t)M * 96;     // [M,256]

    dim3 blk(256);
    // 1-3: independent projections
    gemm_bias_wmma<<<dim3(4, mt), blk, 0, stream>>>(vflat, W_val, b_val, ws_val, M, 256);
    gemm_bias_wmma<<<dim3(3, mt), blk, 0, stream>>>(query, W_off, b_off, ws_off, M, 192);
    gemm_bias_wmma<<<dim3(2, mt), blk, 0, stream>>>(query, W_attn, b_attn, ws_log, M, 96);
    // 4: softmax + bilinear gather + weighted sum
    msda_sample<<<dim3(M), dim3(32, 8), 0, stream>>>(ws_val, ws_off, ws_log, refp, ws_mid);
    // 5: output projection
    gemm_bias_wmma<<<dim3(4, mt), blk, 0, stream>>>(ws_mid, W_out, b_out, out, M, 256);
}
